// PyCorrAssemble_77601469104858
// MI455X (gfx1250) — matrix operand order repaired and checked by reading, not compiled, not run on hardware
//
#include <hip/hip_runtime.h>
#include <hip/hip_bf16.h>
#include <cstdint>
#include <cstddef>

typedef __attribute__((ext_vector_type(16))) _Float16 v16h;
typedef __attribute__((ext_vector_type(8)))  float    v8f;

#define C_FEAT 256
#define NBATCH 8

struct U32x2 { uint4 lo, hi; };

static __device__ __forceinline__ v16h frag_from(uint4 lo, uint4 hi) {
  U32x2 t{lo, hi};
  return __builtin_bit_cast(v16h, t);
}
// 16 contiguous halfs (32B)
static __device__ __forceinline__ v16h ld_contig16(const _Float16* p) {
  return frag_from(*(const uint4*)p, *(const uint4*)(p + 8));
}
// A-fragment: halfs [0..7] at p, [8..15] at p+16 (K split per ISA 7.12.2)
static __device__ __forceinline__ v16h ld_afrag(const _Float16* p) {
  return frag_from(*(const uint4*)p, *(const uint4*)(p + 16));
}

// ============================================================
// Correlation: aff[b, dy*3+dx, y, x] = dot_c(f1[:,y,x], f2[:,y+dy-1,x+dx-1]) / C
// ============================================================
__global__ void corr_kernel(const float* __restrict__ f1,
                            const float* __restrict__ f2,
                            float* __restrict__ aff,
                            int H, int W) {
  int HW = H * W;
  long idx = (long)blockIdx.x * blockDim.x + threadIdx.x;
  long total = (long)NBATCH * HW;
  if (idx >= total) return;
  int x = (int)(idx % W);
  int y = (int)((idx / W) % H);
  int b = (int)(idx / HW);
  const float* p1 = f1 + (size_t)b * C_FEAT * HW + (size_t)y * W + x;
  const float* p2 = f2 + (size_t)b * C_FEAT * HW;
  int  off[9]; bool ok[9];
#pragma unroll
  for (int t = 0; t < 9; ++t) {
    int sy = y + t / 3 - 1, sx = x + t % 3 - 1;
    ok[t]  = ((unsigned)sy < (unsigned)H) && ((unsigned)sx < (unsigned)W);
    off[t] = sy * W + sx;
  }
  float acc[9] = {0.f,0.f,0.f,0.f,0.f,0.f,0.f,0.f,0.f};
  for (int c = 0; c < C_FEAT; ++c) {
    float v = p1[(size_t)c * HW];
    const float* pc = p2 + (size_t)c * HW;
#pragma unroll
    for (int t = 0; t < 9; ++t)
      if (ok[t]) acc[t] = fmaf(v, pc[off[t]], acc[t]);
  }
#pragma unroll
  for (int t = 0; t < 9; ++t)
    aff[((size_t)b * 9 + t) * HW + (size_t)y * W + x] = acc[t] * (1.0f / C_FEAT);
}

// ============================================================
// mix_corr: spatial 2x bilinear upsample of corr_top + window resize
// d_top -> 2*d_top-1 (half-pixel, clamped), *BETA, plus ALPHA * raw-view
// central add of corr_bot. Output NCHW [B, d2*d2, h2, w2].
// ============================================================
__global__ void mix_kernel(const float* __restrict__ bot,
                           const float* __restrict__ top,
                           float* __restrict__ out,
                           int h2, int w2, int d_bot, int d_top) {
  int d2 = 2 * d_top - 1, dd2 = d2 * d2;
  int h = h2 >> 1, w = w2 >> 1;
  long total = (long)NBATCH * dd2 * h2 * w2;
  long idx = (long)blockIdx.x * blockDim.x + threadIdx.x;
  if (idx >= total) return;
  int x  = (int)(idx % w2);
  int y  = (int)((idx / w2) % h2);
  int j  = (int)((idx / ((long)w2 * h2)) % dd2);
  int b  = (int)(idx / ((long)w2 * h2 * dd2));
  int wi = j / d2, wj = j % d2;

  float si = (wi + 0.5f) * (float)d_top / (float)d2 - 0.5f;
  si = fminf(fmaxf(si, 0.0f), (float)(d_top - 1));
  int   i0 = (int)si;  float fi = si - i0;  int i1 = min(i0 + 1, d_top - 1);
  float sj = (wj + 0.5f) * (float)d_top / (float)d2 - 0.5f;
  sj = fminf(fmaxf(sj, 0.0f), (float)(d_top - 1));
  int   j0 = (int)sj;  float fj = sj - j0;  int j1 = min(j0 + 1, d_top - 1);

  float ty = y * 0.5f - 0.25f;
  ty = fminf(fmaxf(ty, 0.0f), (float)(h - 1));
  int y0 = (int)ty; float fy = ty - y0; int y1 = min(y0 + 1, h - 1);
  float tx = x * 0.5f - 0.25f;
  tx = fminf(fmaxf(tx, 0.0f), (float)(w - 1));
  int x0 = (int)tx; float fx = tx - x0; int x1 = min(x0 + 1, w - 1);

  const float* tb = top + (size_t)b * (d_top * d_top) * h * w;
  int   ii[2] = {i0, i1};  float wiw[2] = {1.0f - fi, fi};
  int   jj[2] = {j0, j1};  float wjw[2] = {1.0f - fj, fj};
  float val = 0.0f;
#pragma unroll
  for (int a = 0; a < 2; ++a) {
#pragma unroll
    for (int bb = 0; bb < 2; ++bb) {
      float wwin = wiw[a] * wjw[bb];
      const float* tc = tb + (size_t)(ii[a] * d_top + jj[bb]) * h * w;
      float v = (1.0f - fy) * ((1.0f - fx) * tc[y0 * w + x0] + fx * tc[y0 * w + x1])
              +          fy * ((1.0f - fx) * tc[y1 * w + x0] + fx * tc[y1 * w + x1]);
      val = fmaf(wwin, v, val);
    }
  }
  float o = 0.5f * val;  // BETA
  int c0 = (d2 - d_bot) / 2;
  if (wi >= c0 && wi < c0 + d_bot && wj >= c0 && wj < c0 + d_bot) {
    size_t flat = ((size_t)((size_t)y * w2 + x) * d_bot + (wi - c0)) * d_bot + (wj - c0);
    o += 0.5f * bot[(size_t)b * d_bot * d_bot * h2 * w2 + flat];  // ALPHA, raw .view
  }
  out[idx] = o;
}

// ============================================================
// Pack feat (f32 NCHW) into channel-last X (f16 [B][HW][512], lower 256 ch).
// Thread <-> pixel; strided-coalesced reads, 16B contiguous stores.
// ============================================================
__global__ void pack_feat_kernel(const float* __restrict__ feat,
                                 _Float16* __restrict__ X, int HW) {
  long total = (long)NBATCH * HW;
  long idx = (long)blockIdx.x * blockDim.x + threadIdx.x;
  if (idx >= total) return;
  int b   = (int)(idx / HW);
  int pix = (int)(idx % HW);
  const float* fb = feat + (size_t)b * C_FEAT * HW + pix;
  _Float16* xb = X + ((size_t)b * HW + pix) * 512;
  for (int c0 = 0; c0 < C_FEAT; c0 += 8) {
    _Float16 tmp[8];
#pragma unroll
    for (int i = 0; i < 8; ++i) tmp[i] = (_Float16)fb[(size_t)(c0 + i) * HW];
    *(uint4*)(xb + c0) = *(const uint4*)tmp;
  }
}

__global__ void w1cvt_kernel(const float* __restrict__ w, _Float16* __restrict__ o) {
  int idx = blockIdx.x * blockDim.x + threadIdx.x;
  if (idx < 256 * 512) o[idx] = (_Float16)w[idx];
}

// w2 [16][256][3][3] f32 -> tap-major f16 [9][16][256]
__global__ void w2cvt_kernel(const float* __restrict__ w, _Float16* __restrict__ o) {
  int idx = blockIdx.x * blockDim.x + threadIdx.x;
  if (idx >= 9 * 16 * 256) return;
  int t = idx / (16 * 256);
  int m = (idx / 256) % 16;
  int c = idx % 256;
  o[idx] = (_Float16)w[(size_t)(m * 256 + c) * 9 + t];
}

// ============================================================
// Fused per-pixel softmax over D affinities (LDS-staged) + D-tap assemble;
// writes assembled ref as f16 into upper 256 channels of channel-last X.
// Block = one image row (W threads). Dynamic LDS = D*W floats.
// ============================================================
__global__ void softmax_assemble_kernel(const float* __restrict__ aff,
                                        const float* __restrict__ ref,
                                        _Float16* __restrict__ X,
                                        int D, int k, int H, int W) {
  extern __shared__ float saff[];  // [D][W]
  int x = threadIdx.x;
  int y = blockIdx.x % H;
  int b = blockIdx.x / H;
  int HW = H * W;
  const float* ab = aff + (size_t)b * D * HW + (size_t)y * W + x;
  float mx = -1e30f;
  for (int d = 0; d < D; ++d) mx = fmaxf(mx, ab[(size_t)d * HW]);
  float s = 0.0f;
  for (int d = 0; d < D; ++d) {
    float e = __expf(ab[(size_t)d * HW] - mx);
    saff[d * W + x] = e;
    s += e;
  }
  float inv = 1.0f / s;
  for (int d = 0; d < D; ++d) saff[d * W + x] *= inv;
  __syncthreads();
  const float* rb = ref + (size_t)b * C_FEAT * HW;
  _Float16* xb = X + ((size_t)b * HW + (size_t)y * W + x) * 512 + C_FEAT;
  int win = 2 * k + 1;
  for (int c0 = 0; c0 < C_FEAT; c0 += 8) {
    _Float16 tmp[8];
#pragma unroll
    for (int i = 0; i < 8; ++i) {
      const float* rc = rb + (size_t)(c0 + i) * HW;
      float acc = 0.0f;
      int dbase = 0;
      for (int dy = -k; dy <= k; ++dy) {
        int sy = y + dy;
        if ((unsigned)sy < (unsigned)H) {
          const float* rr = rc + sy * W;
          for (int dx = -k; dx <= k; ++dx) {
            int sx = x + dx;
            if ((unsigned)sx < (unsigned)W)
              acc = fmaf(saff[(dbase + dx + k) * W + x], rr[sx], acc);
          }
        }
        dbase += win;
      }
      tmp[i] = (_Float16)acc;
    }
    *(uint4*)(xb + c0) = *(const uint4*)tmp;
  }
}

// ============================================================
// conv1 (1x1, 512->256) WMMA GEMM, channel-last activations:
//   Out[b][n][m] = sum_k W1[m][k] * X[b][n][k]
// Wave owns one 16-row M tile x two 16-pixel N tiles (A reused, 2 WMMAs/K-step).
// All fragments are 128-bit contiguous loads (K is the fast axis everywhere).
// ============================================================
__global__ void conv1_wmma_kernel(const _Float16* __restrict__ X,   // [B][HW][512]
                                  const _Float16* __restrict__ Wh,  // [256][512]
                                  const float* __restrict__ bias,
                                  _Float16* __restrict__ Y,         // [B][HW][256]
                                  int HW) {
  int lane = threadIdx.x;
  int n0 = blockIdx.x * 32;
  int m0 = blockIdx.y * 16;
  int b  = blockIdx.z;
  int hi  = lane >> 4;
  int l15 = lane & 15;
  const _Float16* arow  = Wh + (size_t)(m0 + l15) * 512 + hi * 16;
  const _Float16* brow0 = X + ((size_t)b * HW + n0 + l15) * 512 + hi * 16;
  const _Float16* brow1 = brow0 + (size_t)16 * 512;
  v8f acc0 = {}, acc1 = {};
  for (int k0 = 0; k0 < 512; k0 += 32) {
    if (k0 + 32 < 512) {
      __builtin_prefetch(brow0 + k0 + 32, 0, 0);
      __builtin_prefetch(brow1 + k0 + 32, 0, 0);
    }
    v16h a   = ld_contig16(arow + k0);
    v16h bf0 = ld_contig16(brow0 + k0);
    v16h bf1 = ld_contig16(brow1 + k0);
    acc0 = __builtin_amdgcn_wmma_f32_16x16x32_f16(false, a, false, bf0,
                                                  (short)0, acc0, false, false);
    acc1 = __builtin_amdgcn_wmma_f32_16x16x32_f16(false, a, false, bf1,
                                                  (short)0, acc1, false, false);
  }
  // epilogue: 8 contiguous halfs per lane -> one 16B store per tile
  {
    _Float16 tmp[8];
#pragma unroll
    for (int r = 0; r < 8; ++r) tmp[r] = (_Float16)(acc0[r] + bias[m0 + hi * 8 + r]);
    _Float16* yp = Y + ((size_t)b * HW + n0 + l15) * 256 + m0 + hi * 8;
    *(uint4*)yp = *(const uint4*)tmp;
  }
  {
    _Float16 tmp[8];
#pragma unroll
    for (int r = 0; r < 8; ++r) tmp[r] = (_Float16)(acc1[r] + bias[m0 + hi * 8 + r]);
    _Float16* yp = Y + ((size_t)b * HW + n0 + 16 + l15) * 256 + m0 + hi * 8;
    *(uint4*)yp = *(const uint4*)tmp;
  }
}

// ============================================================
// conv2 (3x3, 256->16) as 9 shifted 1x1 WMMA GEMMs (channel-last input).
// M=16 == one WMMA tile; zero-pad via per-lane zero fragments (reconverged
// before WMMA issue, EXEC all-ones).
// ============================================================
__global__ void conv2_wmma_kernel(const _Float16* __restrict__ Y1,  // [B][HW][256]
                                  const _Float16* __restrict__ W2h, // [9][16][256]
                                  const float* __restrict__ b2,
                                  float* __restrict__ Y2,           // [B][16][HW]
                                  int H, int W) {
  int lane = threadIdx.x;
  int HW = H * W;
  int n0 = blockIdx.x * 16;
  int b  = blockIdx.z;
  int hi  = lane >> 4;
  int l15 = lane & 15;
  int n  = n0 + l15;
  int py = n / W, px = n % W;
  const _Float16* Yb = Y1 + (size_t)b * HW * 256;
  v8f acc = {};
#pragma unroll
  for (int t = 0; t < 9; ++t) {
    int sy = py + t / 3 - 1, sx = px + t % 3 - 1;
    bool ok = ((unsigned)sy < (unsigned)H) && ((unsigned)sx < (unsigned)W);
    const _Float16* brow = Yb + (size_t)(sy * W + sx) * 256 + hi * 16;
    const _Float16* arow = W2h + (size_t)t * 16 * 256 + (size_t)l15 * 256 + hi * 16;
    for (int k0 = 0; k0 < 256; k0 += 32) {
      v16h a = ld_contig16(arow + k0);
      uint4 blo = {0u, 0u, 0u, 0u}, bhi = {0u, 0u, 0u, 0u};
      if (ok) {
        blo = *(const uint4*)(brow + k0);
        bhi = *(const uint4*)(brow + k0 + 8);
      }
      v16h bf = frag_from(blo, bhi);
      acc = __builtin_amdgcn_wmma_f32_16x16x32_f16(false, a, false, bf,
                                                   (short)0, acc, false, false);
    }
  }
  float* Ob = Y2 + (size_t)b * 16 * HW;
#pragma unroll
  for (int r = 0; r < 8; ++r) {
    int m = r + hi * 8;
    Ob[(size_t)m * HW + n] = acc[r] + b2[m];
  }
}

// ============================================================
// conv3 (3x3, 16->3), plain fp32 VALU (tiny)
// ============================================================
__global__ void conv3_kernel(const float* __restrict__ Y2,
                             const float* __restrict__ w3,
                             const float* __restrict__ b3,
                             float* __restrict__ Y3, int H, int W) {
  int HW = H * W;
  long total = (long)NBATCH * HW;
  long idx = (long)blockIdx.x * blockDim.x + threadIdx.x;
  if (idx >= total) return;
  int x = (int)(idx % W);
  int y = (int)((idx / W) % H);
  int b = (int)(idx / HW);
  const float* Ib = Y2 + (size_t)b * 16 * HW;
  float o0 = b3[0], o1 = b3[1], o2 = b3[2];
  for (int c = 0; c < 16; ++c) {
    const float* Ic = Ib + (size_t)c * HW;
#pragma unroll
    for (int t = 0; t < 9; ++t) {
      int sy = y + t / 3 - 1, sx = x + t % 3 - 1;
      if ((unsigned)sy < (unsigned)H && (unsigned)sx < (unsigned)W) {
        float v = Ic[sy * W + sx];
        o0 = fmaf(v, w3[(0 * 16 + c) * 9 + t], o0);
        o1 = fmaf(v, w3[(1 * 16 + c) * 9 + t], o1);
        o2 = fmaf(v, w3[(2 * 16 + c) * 9 + t], o2);
      }
    }
  }
  size_t p = (size_t)y * W + x;
  Y3[((size_t)b * 3 + 0) * HW + p] = o0;
  Y3[((size_t)b * 3 + 1) * HW + p] = o1;
  Y3[((size_t)b * 3 + 2) * HW + p] = o2;
}

// ============================================================
// conv4 (3x3, 3->2) + channel softmax -> blend scores S[b][2][HW]
// ============================================================
__global__ void conv4_softmax_kernel(const float* __restrict__ Y3,
                                     const float* __restrict__ w4,
                                     const float* __restrict__ b4,
                                     float* __restrict__ S, int H, int W) {
  int HW = H * W;
  long total = (long)NBATCH * HW;
  long idx = (long)blockIdx.x * blockDim.x + threadIdx.x;
  if (idx >= total) return;
  int x = (int)(idx % W);
  int y = (int)((idx / W) % H);
  int b = (int)(idx / HW);
  float l0 = b4[0], l1 = b4[1];
  for (int c = 0; c < 3; ++c) {
    const float* Ic = Y3 + ((size_t)b * 3 + c) * HW;
#pragma unroll
    for (int t = 0; t < 9; ++t) {
      int sy = y + t / 3 - 1, sx = x + t % 3 - 1;
      if ((unsigned)sy < (unsigned)H && (unsigned)sx < (unsigned)W) {
        float v = Ic[sy * W + sx];
        l0 = fmaf(v, w4[(0 * 3 + c) * 9 + t], l0);
        l1 = fmaf(v, w4[(1 * 3 + c) * 9 + t], l1);
      }
    }
  }
  float m  = fmaxf(l0, l1);
  float e0 = __expf(l0 - m), e1 = __expf(l1 - m);
  float inv = 1.0f / (e0 + e1);
  size_t p = (size_t)y * W + x;
  S[(size_t)b * 2 * HW + p]      = e0 * inv;
  S[(size_t)b * 2 * HW + HW + p] = e1 * inv;
}

// ============================================================
// out = s0*feat + s1*upd   (upd = upper 256 channels of channel-last X)
// ============================================================
__global__ void blend_kernel(const float* __restrict__ feat,
                             const _Float16* __restrict__ X,
                             const float* __restrict__ S,
                             float* __restrict__ out, int HW) {
  long chw = (long)C_FEAT * HW;
  long total = (long)NBATCH * chw;
  long idx = (long)blockIdx.x * blockDim.x + threadIdx.x;
  if (idx >= total) return;
  int b = (int)(idx / chw);
  long r = idx % chw;
  int c   = (int)(r / HW);
  int pix = (int)(r % HW);
  float s0 = S[(size_t)b * 2 * HW + pix];
  float s1 = S[(size_t)b * 2 * HW + HW + pix];
  float upd = (float)X[((size_t)b * HW + pix) * 512 + C_FEAT + c];
  out[idx] = fmaf(s0, feat[idx], s1 * upd);
}

// ============================================================
extern "C" void kernel_launch(void* const* d_in, const int* in_sizes, int n_in,
                              void* d_out, int out_size, void* d_ws, size_t ws_size,
                              hipStream_t stream) {
  (void)in_sizes; (void)n_in; (void)out_size; (void)ws_size;
  const float* feat[3] = {(const float*)d_in[0], (const float*)d_in[1], (const float*)d_in[2]};
  const float* ref [3] = {(const float*)d_in[5], (const float*)d_in[6], (const float*)d_in[7]};
  const int  Hs[3]  = {128, 64, 32};
  const long HWs[3] = {16384, 4096, 1024};

  const float *w1[3], *b1[3], *w2[3], *b2[3], *w3[3], *b3[3], *w4[3], *b4[3];
  for (int s = 0; s < 3; ++s) {
    int base = 10 + 8 * s;
    w1[s] = (const float*)d_in[base + 0];  b1[s] = (const float*)d_in[base + 1];
    w2[s] = (const float*)d_in[base + 2];  b2[s] = (const float*)d_in[base + 3];
    w3[s] = (const float*)d_in[base + 4];  b3[s] = (const float*)d_in[base + 5];
    w4[s] = (const float*)d_in[base + 6];  b4[s] = (const float*)d_in[base + 7];
  }

  // ---- workspace carve-up (bump allocator, 256B aligned) ----
  char* ws = (char*)d_ws;
  size_t off = 0;
  auto alloc = [&](size_t bytes) -> char* {
    char* p = ws + off;
    off = (off + bytes + 255) & ~(size_t)255;
    return p;
  };
  float* aff2  = (float*)alloc((size_t)8 * 9  * 1024  * 4);
  float* aff1r = (float*)alloc((size_t)8 * 9  * 4096  * 4);
  float* aff0r = (float*)alloc((size_t)8 * 9  * 16384 * 4);
  float* aff1m = (float*)alloc((size_t)8 * 25 * 4096  * 4);
  float* aff0m = (float*)alloc((size_t)8 * 81 * 16384 * 4);
  _Float16* X[3]; _Float16* Y1[3]; float* Y2[3]; float* Y3[3]; float* S[3];
  _Float16* W1h[3]; _Float16* W2h[3];
  for (int s = 0; s < 3; ++s) {
    X [s] = (_Float16*)alloc((size_t)8 * 512 * HWs[s] * 2);
    Y1[s] = (_Float16*)alloc((size_t)8 * 256 * HWs[s] * 2);
    Y2[s] = (float*)   alloc((size_t)8 * 16  * HWs[s] * 4);
    Y3[s] = (float*)   alloc((size_t)8 * 3   * HWs[s] * 4);
    S [s] = (float*)   alloc((size_t)8 * 2   * HWs[s] * 4);
    W1h[s] = (_Float16*)alloc((size_t)256 * 512 * 2);
    W2h[s] = (_Float16*)alloc((size_t)9 * 16 * 256 * 2);
  }

  // ---- weight conversion (f32 -> f16, conv2 tap-major) ----
  for (int s = 0; s < 3; ++s) {
    w1cvt_kernel<<<(256 * 512 + 255) / 256, 256, 0, stream>>>(w1[s], W1h[s]);
    w2cvt_kernel<<<(9 * 16 * 256 + 255) / 256, 256, 0, stream>>>(w2[s], W2h[s]);
  }

  // ---- correlations ----
  { long t = 8L * 1024;  corr_kernel<<<(unsigned)((t + 255) / 256), 256, 0, stream>>>(feat[2], ref[2], aff2,  32, 32); }
  { long t = 8L * 4096;  corr_kernel<<<(unsigned)((t + 255) / 256), 256, 0, stream>>>(feat[1], ref[1], aff1r, 64, 64); }
  { long t = 8L * 16384; corr_kernel<<<(unsigned)((t + 255) / 256), 256, 0, stream>>>(feat[0], ref[0], aff0r, 128, 128); }

  // ---- pyramid mixing ----
  { long t = 8L * 25 * 4096;  mix_kernel<<<(unsigned)((t + 255) / 256), 256, 0, stream>>>(aff1r, aff2,  aff1m, 64, 64, 3, 3); }
  { long t = 8L * 81 * 16384; mix_kernel<<<(unsigned)((t + 255) / 256), 256, 0, stream>>>(aff0r, aff1m, aff0m, 128, 128, 3, 5); }

  // ---- pack feat into lower 256 channels of channel-last X (f16) ----
  for (int s = 0; s < 3; ++s) {
    long t = 8L * HWs[s];
    pack_feat_kernel<<<(unsigned)((t + 255) / 256), 256, 0, stream>>>(feat[s], X[s], (int)HWs[s]);
  }

  // ---- fused softmax + assemble -> upper 256 channels of X (f16) ----
  softmax_assemble_kernel<<<8 * 128, 128, 81 * 128 * 4, stream>>>(aff0m, ref[0],  X[0], 81, 4, 128, 128);
  softmax_assemble_kernel<<<8 * 64,  64,  25 * 64  * 4, stream>>>(aff1m, ref[1],  X[1], 25, 2, 64, 64);
  softmax_assemble_kernel<<<8 * 32,  32,  9  * 32  * 4, stream>>>(aff2,  feat[2], X[2], 9, 1, 32, 32);  // NB: reference uses feat2 here

  // ---- conv stack per scale ----
  for (int s = 0; s < 3; ++s) {
    dim3 g1((unsigned)(HWs[s] / 32), 16, 8);
    conv1_wmma_kernel<<<g1, 32, 0, stream>>>(X[s], W1h[s], b1[s], Y1[s], (int)HWs[s]);
    dim3 g2((unsigned)(HWs[s] / 16), 1, 8);
    conv2_wmma_kernel<<<g2, 32, 0, stream>>>(Y1[s], W2h[s], b2[s], Y2[s], Hs[s], Hs[s]);
    long t3 = 8L * HWs[s];
    conv3_kernel<<<(unsigned)((t3 + 255) / 256), 256, 0, stream>>>(Y2[s], w3[s], b3[s], Y3[s], Hs[s], Hs[s]);
    conv4_softmax_kernel<<<(unsigned)((t3 + 255) / 256), 256, 0, stream>>>(Y3[s], w4[s], b4[s], S[s], Hs[s], Hs[s]);
  }

  // ---- blend into d_out ----
  float* out = (float*)d_out;
  size_t ooff = 0;
  for (int s = 0; s < 3; ++s) {
    long t = 8L * 256 * HWs[s];
    blend_kernel<<<(unsigned)((t + 255) / 256), 256, 0, stream>>>(feat[s], X[s], S[s], out + ooff, (int)HWs[s]);
    ooff += (size_t)t;
  }

  // ---- passthrough feat3 / feat4 ----
  hipMemcpyAsync(out + ooff, d_in[3], (size_t)8 * 256 * 16 * 16 * 4, hipMemcpyDeviceToDevice, stream);
  ooff += (size_t)8 * 256 * 16 * 16;
  hipMemcpyAsync(out + ooff, d_in[4], (size_t)8 * 256 * 8 * 8 * 4, hipMemcpyDeviceToDevice, stream);
}
